// VectorQuantizer_88227218194964
// MI455X (gfx1250) — matrix operand order repaired and checked by reading, not compiled, not run on hardware
//
#include <hip/hip_runtime.h>

typedef __attribute__((ext_vector_type(16))) __bf16 v16bf;
typedef __attribute__((ext_vector_type(8)))  __bf16 v8bf;
typedef __attribute__((ext_vector_type(8)))  float  v8f;

#define MROWS  16384   // 16*1024 flattened z rows
#define KCODES 8192    // codebook entries
#define CDIM   256     // channels
#define MB     64      // rows per block (4 waves x 16 rows)
#define LDSPAD 264     // padded row stride (bf16 elems): 528 B = 33*16, banks spread
#define NSTAGE (KCODES / 16)   // 512 stages of 16 codes

static __device__ __forceinline__ v8f wmma_bf16f32(v16bf a, v16bf b, v8f c) {
  // D = A(16x32 bf16) * B(32x16 bf16) + C(16x16 f32)
  return __builtin_amdgcn_wmma_f32_16x16x32_bf16(false, a, false, b, (short)0, c,
                                                 false, false);
}

// ---------------------------------------------------------------------------
// Kernel 0: per codebook row compute ||e||^2 AND pre-split bf16 hi/lo copies.
// One wave32 per row; 8 rows per 256-thread block.
// ---------------------------------------------------------------------------
__global__ void __launch_bounds__(256)
vq_prep_kernel(const float* __restrict__ emb, float* __restrict__ enorm,
               __bf16* __restrict__ ghi, __bf16* __restrict__ glo) {
  const int wave = threadIdx.x >> 5;
  const int lane = threadIdx.x & 31;
  const int row  = blockIdx.x * 8 + wave;
  const float* e = emb + (size_t)row * CDIM;
  float s = 0.f;
  #pragma unroll
  for (int i = 0; i < 8; ++i) {
    const int idx = lane + i * 32;
    const float v = e[idx];
    s += v * v;
    const __bf16 hi = (__bf16)v;
    ghi[(size_t)row * CDIM + idx] = hi;
    glo[(size_t)row * CDIM + idx] = (__bf16)(v - (float)hi);
  }
  #pragma unroll
  for (int off = 16; off >= 1; off >>= 1) s += __shfl_xor(s, off, 32);
  if (lane == 0) enorm[row] = s;
}

// ---------------------------------------------------------------------------
// Async stage fetch: 16 codebook rows (hi + lo bf16) -> LDS buffer.
// 16 rows x 256 cols x 2B = 8 KB per array = 512 b128 chunks per array;
// 128 threads x 4 chunks each; 8 async instructions per wave per stage.
// ---------------------------------------------------------------------------
static __device__ __forceinline__ void
issue_async_stage(unsigned long long ghi_a, unsigned long long glo_a, int stage,
                  __bf16* ldshi, __bf16* ldslo, int tid) {
  #pragma unroll
  for (int i = 0; i < 4; ++i) {
    const int c    = i * 128 + tid;
    const int row  = c >> 5;          // 0..15
    const int col8 = (c & 31) << 3;   // 0,8,...,248 (bf16 elems)
    const unsigned goff = (unsigned)((((stage << 4) + row) * CDIM + col8) * 2);
    const unsigned lh = (unsigned)(unsigned long long)(ldshi + row * LDSPAD + col8);
    const unsigned ll = (unsigned)(unsigned long long)(ldslo + row * LDSPAD + col8);
    asm volatile("global_load_async_to_lds_b128 %0, %1, %2"
                 :: "v"(lh), "v"(goff), "s"(ghi_a) : "memory");
    asm volatile("global_load_async_to_lds_b128 %0, %1, %2"
                 :: "v"(ll), "v"(goff), "s"(glo_a) : "memory");
  }
}

// ---------------------------------------------------------------------------
// Kernel 1: fused (z . e^T) via bf16 hi/lo split WMMA + running argmin.
// 128 threads (4 waves); each wave owns 16 z-rows and scans all 8192 codes.
// Double-buffered LDS stages prefetched 2 ahead with async global->LDS.
// Three independent WMMA accumulator chains (hi*hi / hi*lo / lo*hi).
// ---------------------------------------------------------------------------
__global__ void __launch_bounds__(128)
vq_argmin_kernel(const float* __restrict__ z, const __bf16* __restrict__ ghi,
                 const __bf16* __restrict__ glo, const float* __restrict__ enorm,
                 int* __restrict__ bestIdx) {
  // [buffer][hi|lo][row][col]
  __shared__ __attribute__((aligned(16))) __bf16 lds[2][2][16][LDSPAD];

  const int tid  = threadIdx.x;
  const int w    = tid >> 5;
  const int lane = tid & 31;
  const int r    = lane & 15;       // A row / B column within tile
  const int h    = lane >> 4;       // half-group (K sub-range select)
  const int mrow = blockIdx.x * MB + w * 16;

  const unsigned long long ghi_a = (unsigned long long)ghi;
  const unsigned long long glo_a = (unsigned long long)glo;

  // ---- Preload this wave's 16x256 A tile (z rows) as bf16 hi/lo fragments.
  // A layout (16-bit, 16x32): lane r holds row r; frag elems 0..7 -> K=h*8+0..7,
  // elems 8..15 -> K=16+h*8+0..7.
  const float* zrow = z + (size_t)(mrow + r) * CDIM;
  v16bf a_hi[8], a_lo[8];
  #pragma unroll
  for (int ks = 0; ks < 8; ++ks) {
    float xs[16];
    const float4 c0 = *(const float4*)(zrow + ks * 32 + h * 8);
    const float4 c1 = *(const float4*)(zrow + ks * 32 + h * 8 + 4);
    const float4 c2 = *(const float4*)(zrow + ks * 32 + 16 + h * 8);
    const float4 c3 = *(const float4*)(zrow + ks * 32 + 16 + h * 8 + 4);
    xs[0]=c0.x;  xs[1]=c0.y;  xs[2]=c0.z;  xs[3]=c0.w;
    xs[4]=c1.x;  xs[5]=c1.y;  xs[6]=c1.z;  xs[7]=c1.w;
    xs[8]=c2.x;  xs[9]=c2.y;  xs[10]=c2.z; xs[11]=c2.w;
    xs[12]=c3.x; xs[13]=c3.y; xs[14]=c3.z; xs[15]=c3.w;
    #pragma unroll
    for (int e2 = 0; e2 < 16; ++e2) {
      const __bf16 hi = (__bf16)xs[e2];
      a_hi[ks][e2] = hi;
      a_lo[ks][e2] = (__bf16)(xs[e2] - (float)hi);
    }
  }

  float best[8];
  int   bidx[8];
  #pragma unroll
  for (int j = 0; j < 8; ++j) { best[j] = 3.0e38f; bidx[j] = 0; }

  // Prologue: prefetch stages 0 and 1.
  issue_async_stage(ghi_a, glo_a, 0, &lds[0][0][0][0], &lds[0][1][0][0], tid);
  issue_async_stage(ghi_a, glo_a, 1, &lds[1][0][0][0], &lds[1][1][0][0], tid);

  for (int s = 0; s < NSTAGE; ++s) {
    const int buf = s & 1;
    // Only the most recent batch (8 async ops) may remain outstanding ->
    // stage s's data has landed in LDS (async loads complete in order).
    asm volatile("s_wait_asynccnt 8" ::: "memory");
    __syncthreads();

    const __bf16* bh_base = &lds[buf][0][0][0];
    const __bf16* bl_base = &lds[buf][1][0][0];
    v8f acc0 = {0.f, 0.f, 0.f, 0.f, 0.f, 0.f, 0.f, 0.f};  // hi*hi
    v8f acc1 = {0.f, 0.f, 0.f, 0.f, 0.f, 0.f, 0.f, 0.f};  // hi*lo
    v8f acc2 = {0.f, 0.f, 0.f, 0.f, 0.f, 0.f, 0.f, 0.f};  // lo*hi
    #pragma unroll
    for (int ks = 0; ks < 8; ++ks) {
      // B layout (16-bit, 32x16): lane holds column r; elems 0..7 -> K=h*8+0..7, elems 8..15 -> K=16+h*8+0..7.
      const int kb = ks * 32 + h * 8;
      const __bf16* ph = bh_base + r * LDSPAD + kb;
      const __bf16* pl = bl_base + r * LDSPAD + kb;
      v8bf p0 = *(const v8bf*)(ph);
      v8bf p1 = *(const v8bf*)(ph + 16);
      v16bf bh = __builtin_shufflevector(p0, p1, 0,1,2,3,4,5,6,7,
                                                 8,9,10,11,12,13,14,15);
      v8bf q0 = *(const v8bf*)(pl);
      v8bf q1 = *(const v8bf*)(pl + 16);
      v16bf bl = __builtin_shufflevector(q0, q1, 0,1,2,3,4,5,6,7,
                                                 8,9,10,11,12,13,14,15);
      // hi*hi + hi*lo + lo*hi ~= fp32 dot product; 3 independent chains
      acc0 = wmma_bf16f32(a_hi[ks], bh, acc0);
      acc1 = wmma_bf16f32(a_hi[ks], bl, acc1);
      acc2 = wmma_bf16f32(a_lo[ks], bh, acc2);
    }

    // All LDS reads for this stage have completed (WMMA consumed them) ->
    // release the buffer and start prefetching stage s+2 into it.
    __syncthreads();
    if (s + 2 < NSTAGE)
      issue_async_stage(ghi_a, glo_a, s + 2,
                        &lds[buf][0][0][0], &lds[buf][1][0][0], tid);

    const int   ng = (s << 4) + r;
    const float en = enorm[ng];
    // C/D layout: lane holds column n=r; VGPR j holds row h*8 + j.
    #pragma unroll
    for (int j = 0; j < 8; ++j) {
      const float dot = acc0[j] + (acc1[j] + acc2[j]);
      const float sc  = en - 2.0f * dot;   // ||e||^2 - 2 x.e
      if (sc < best[j]) { best[j] = sc; bidx[j] = ng; }
    }
  }

  // Argmin across the 16 lanes of each half group (tie -> lower index).
  #pragma unroll
  for (int j = 0; j < 8; ++j) {
    float s2 = best[j];
    int   bi = bidx[j];
    #pragma unroll
    for (int off = 1; off < 16; off <<= 1) {
      const float so = __shfl_xor(s2, off, 32);
      const int   io = __shfl_xor(bi, off, 32);
      if (so < s2 || (so == s2 && io < bi)) { s2 = so; bi = io; }
    }
    if (r == 0) bestIdx[mrow + h * 8 + j] = bi;
  }
}

// ---------------------------------------------------------------------------
// Kernel 2: gather codebook rows, write quantized_ste, indices (as float),
// and per-block partial sums of (q - z)^2. 16 rows per 256-thread block.
// ---------------------------------------------------------------------------
__global__ void __launch_bounds__(256)
vq_gather_loss_kernel(const float* __restrict__ z, const float* __restrict__ emb,
                      const int* __restrict__ bidx, float* __restrict__ outq,
                      float* __restrict__ outidx, float* __restrict__ partials) {
  const int t = threadIdx.x;
  float accum = 0.f;
  #pragma unroll
  for (int rr = 0; rr < 16; ++rr) {
    const int row = blockIdx.x * 16 + rr;
    const int ci  = bidx[row];
    const float e  = emb[(size_t)ci * CDIM + t];
    const float zz = z[(size_t)row * CDIM + t];
    const float d  = e - zz;
    outq[(size_t)row * CDIM + t] = zz + d;  // z + (q - z): reference STE rounding
    accum += d * d;
  }
  if (t < 16) {
    const int row = blockIdx.x * 16 + t;
    outidx[row] = (float)bidx[row];
  }
  __shared__ float red[256];
  red[t] = accum;
  __syncthreads();
  #pragma unroll
  for (int s = 128; s >= 32; s >>= 1) {
    if (t < s) red[t] += red[t + s];
    __syncthreads();
  }
  if (t < 32) {
    float v = red[t];
    #pragma unroll
    for (int off = 16; off >= 1; off >>= 1) v += __shfl_xor(v, off, 32);
    if (t == 0) partials[blockIdx.x] = v;
  }
}

// ---------------------------------------------------------------------------
// Kernel 3: reduce 1024 partials -> vq_loss = 1.25 * MSE (q==e => both MSEs equal).
// ---------------------------------------------------------------------------
__global__ void vq_finalize_kernel(const float* __restrict__ partials,
                                   float* __restrict__ out_loss) {
  const int t = threadIdx.x;
  float s = partials[t] + partials[t + 256] + partials[t + 512] + partials[t + 768];
  __shared__ float red[256];
  red[t] = s;
  __syncthreads();
  #pragma unroll
  for (int st = 128; st >= 32; st >>= 1) {
    if (t < st) red[t] += red[t + st];
    __syncthreads();
  }
  if (t < 32) {
    float v = red[t];
    #pragma unroll
    for (int off = 16; off >= 1; off >>= 1) v += __shfl_xor(v, off, 32);
    if (t == 0)
      out_loss[0] = 1.25f * v / (float)((size_t)MROWS * CDIM);
  }
}

// ---------------------------------------------------------------------------
extern "C" void kernel_launch(void* const* d_in, const int* in_sizes, int n_in,
                              void* d_out, int out_size, void* d_ws, size_t ws_size,
                              hipStream_t stream) {
  (void)in_sizes; (void)n_in; (void)out_size; (void)ws_size;
  const float* z   = (const float*)d_in[0];     // (16,1024,256) f32
  const float* emb = (const float*)d_in[1];     // (8192,256) f32
  float* out = (float*)d_out;
  // d_out layout: quantized_ste (16*1024*256) | vq_loss (1) | indices (16384)
  float* out_q    = out;
  float* out_loss = out + (size_t)MROWS * CDIM;
  float* out_idx  = out + (size_t)MROWS * CDIM + 1;

  // workspace: ghi (4MB bf16) | glo (4MB bf16) | enorm f32 | bestIdx i32 | partials f32
  __bf16* ghi     = (__bf16*)d_ws;
  __bf16* glo     = ghi + (size_t)KCODES * CDIM;
  char*   tail    = (char*)(glo + (size_t)KCODES * CDIM);
  float*  enorm   = (float*)tail;
  int*    bestIdx = (int*)(tail + (size_t)KCODES * 4);
  float*  partials= (float*)(tail + (size_t)KCODES * 4 + (size_t)MROWS * 4);

  vq_prep_kernel<<<KCODES / 8, 256, 0, stream>>>(emb, enorm, ghi, glo);
  vq_argmin_kernel<<<MROWS / MB, 128, 0, stream>>>(z, ghi, glo, enorm, bestIdx);
  vq_gather_loss_kernel<<<MROWS / 16, 256, 0, stream>>>(z, emb, bestIdx, out_q,
                                                        out_idx, partials);
  vq_finalize_kernel<<<1, 256, 0, stream>>>(partials, out_loss);
}
